// up_block_26792005992605
// MI455X (gfx1250) — matrix operand order repaired and checked by reading, not compiled, not run on hardware
//
#include <hip/hip_runtime.h>

typedef __attribute__((ext_vector_type(16))) __bf16 v16bf;
typedef __attribute__((ext_vector_type(8)))  float  v8f;
typedef __attribute__((ext_vector_type(4)))  unsigned int u32x4;
typedef __attribute__((ext_vector_type(8)))  unsigned int u32x8;

#define IN_CH 64
#define OUT_CH 32
#define V1 40962
#define V2 163842
#define NBATCH 4
#define EPS 1e-5f
#define SLOPE 0.2f

#define MT1 2561                 /* ceil(V1/16) */
#define MT2 10241                /* ceil(V2/16) */
#define V1PAD (MT1 * 16)         /* raw rows padded to tile multiple */
#define V2PAD (MT2 * 16)         /* z rows padded to tile multiple */

// 16-bit A-matrix (16xK=32) lane/element -> K mapping (ISA 7.12.2)
__device__ __forceinline__ int a_kmap(int e, int g) {
    int v = e >> 1, h = e & 1;
    return ((v >> 2) << 4) + ((v & 3) << 1) + h + (g << 3);
}
// 16-bit B-matrix (K=32 x 16) lane/element -> K mapping
__device__ __forceinline__ int b_kmap(int e, int g) {
    int v = e >> 1, h = e & 1;
    return ((v >> 2) << 4) + ((v & 3) << 1) + h + (g << 3);
}

// TDM: stage `bytes` (multiple of 4, tile_dim0 fits 16 bits) global -> LDS.
__device__ __forceinline__ void tdm_stage_to_lds(const void* gsrc, void* ldst,
                                                 unsigned int bytes) {
    unsigned long long ga = (unsigned long long)gsrc;
    unsigned int ldsoff = (unsigned int)(size_t)ldst;  // flat LDS addr low bits
    unsigned int ndw = bytes >> 2;
    u32x4 g0;
    g0[0] = 1u;                                   // count=1, user mode
    g0[1] = ldsoff;                               // lds_addr
    g0[2] = (unsigned int)ga;                     // global_addr[31:0]
    g0[3] = ((unsigned int)(ga >> 32) & 0x01ffffffu) | (2u << 30);  // addr[56:32] | type=2
    u32x8 g1;
    g1[0] = (2u << 16);                           // wg_mask=0, data_size=4B
    g1[1] = (ndw & 0xffffu) << 16;                // tensor_dim0[15:0]
    g1[2] = (ndw >> 16) | (1u << 16);             // tensor_dim0[31:16] | tensor_dim1=1
    g1[3] = (ndw & 0xffffu) << 16;                // tile_dim0 = ndw
    g1[4] = 1u;                                   // tile_dim1=1, tile_dim2=0
    g1[5] = ndw;                                  // tensor_dim0_stride[31:0]
    g1[6] = 0u;
    g1[7] = 0u;
    asm volatile("tensor_load_to_lds %0, %1" :: "s"(g0), "s"(g1) : "memory");
    __builtin_amdgcn_s_wait_tensorcnt(0);
}

// Async copy: 16B LDS -> global via the async engine (no VGPR round trip).
// Caller must ensure prior ds stores completed (s_wait_dscnt 0).
__device__ __forceinline__ void async_store_b128(void* gdst, const void* lsrc) {
    unsigned long long ga = (unsigned long long)gdst;
    unsigned int la = (unsigned int)(size_t)lsrc;
    asm volatile("global_store_async_from_lds_b128 %0, %1, off" :: "v"(ga), "v"(la) : "memory");
}
__device__ __forceinline__ void wait_ds_done() {
    asm volatile("s_wait_dscnt 0x0" ::: "memory");
}

// ---------------- weight swizzle: f32 (N x Ktot) -> bf16 WMMA B-operand tiles ----------------
__global__ void swizzle_weights(const float* __restrict__ W, __bf16* __restrict__ SW,
                                int NT, int KC, int Ktot) {
    int t = blockIdx.x * blockDim.x + threadIdx.x;
    if (t >= NT * KC * 32) return;
    int lane = t & 31;
    int kc = (t >> 5) % KC;
    int nt = (t >> 5) / KC;
    int n = lane & 15, g = lane >> 4;
    __bf16* dst = SW + ((size_t)(nt * KC + kc) * 32 + lane) * 16;
#pragma unroll
    for (int e = 0; e < 16; ++e) {
        int k = kc * 32 + b_kmap(e, g);
        dst[e] = (__bf16)W[(size_t)(nt * 16 + n) * Ktot + k];
    }
}

__global__ void zero_stats(float* p, int n) {
    int t = blockIdx.x * blockDim.x + threadIdx.x;
    if (t < n) p[t] = 0.f;
}

// ---------------- upconv GEMM: raw[b, v, 0:224] = x1[b,:,v]^T @ W_up^T + b_up ----------------
__global__ void upconv_gemm(const float* __restrict__ x1, const __bf16* __restrict__ SWup,
                            const float* __restrict__ b_up, __bf16* __restrict__ raw) {
    __shared__ __align__(32) __bf16 stg[8][16 * 224];   // 7168 B per wave
    const int b = blockIdx.y;
    const int lane = threadIdx.x & 31;
    const int wave = threadIdx.x >> 5;
    const int tile = blockIdx.x * (blockDim.x >> 5) + wave;
    if (tile >= MT1) return;
    const int m = lane & 15, g = lane >> 4;
    const int vert = tile * 16 + m;
    const int vc = vert < V1 ? vert : V1 - 1;
    const float* x1b = x1 + (size_t)b * IN_CH * V1;

    v16bf A[2];
#pragma unroll
    for (int kc = 0; kc < 2; ++kc)
#pragma unroll
        for (int e = 0; e < 16; ++e) {
            int k = kc * 32 + a_kmap(e, g);
            A[kc][e] = (__bf16)x1b[(size_t)k * V1 + vc];
        }

    __bf16* slab = stg[wave];
#pragma unroll
    for (int nt = 0; nt < 14; ++nt) {
        v8f acc = {};
#pragma unroll
        for (int kc = 0; kc < 2; ++kc) {
            v16bf Bt = *(const v16bf*)(SWup + ((size_t)(nt * 2 + kc) * 32 + lane) * 16);
            acc = __builtin_amdgcn_wmma_f32_16x16x32_bf16(false, A[kc], false, Bt,
                                                          (short)0, acc, false, false);
        }
        float bias = b_up[nt * 16 + m];
#pragma unroll
        for (int r = 0; r < 8; ++r)
            slab[(r + 8 * g) * 224 + nt * 16 + m] = (__bf16)(acc[r] + bias);
    }
    // contiguous 7168B tile -> global: async LDS->global, 224B (14 x b128) per lane
    wait_ds_done();
    __bf16* dst = raw + ((size_t)b * V1PAD + (size_t)tile * 16) * 224;
#pragma unroll
    for (int i = 0; i < 14; ++i)
        async_store_b128(dst + lane * 112 + i * 8, slab + lane * 112 + i * 8);
}

// ---------------- build y (V2 x 64 bf16): upconv scatter + skip concat ----------------
__global__ void build_y(const __bf16* __restrict__ raw, const float* __restrict__ x2,
                        const int* __restrict__ top, const int* __restrict__ down,
                        __bf16* __restrict__ y) {
    size_t idx = (size_t)blockIdx.x * blockDim.x + threadIdx.x;
    if (idx >= (size_t)NBATCH * V2) return;
    int b = (int)(idx / V2);
    int v = (int)(idx % V2);
    const __bf16* rawb = raw + (size_t)b * V1PAD * 224;
    __bf16* yr = y + idx * 64;

    unsigned int w[32];  // 64 bf16 packed as dwords
    if (v < V1) {
        const unsigned int* r0 = (const unsigned int*)(rawb + (size_t)top[v] * 32);
#pragma unroll
        for (int c = 0; c < 16; ++c) w[c] = r0[c];
    } else {
        int i = v - V1;
        const unsigned int* r0 = (const unsigned int*)(rawb + (size_t)down[2 * i] * 32);
        const unsigned int* r1 = (const unsigned int*)(rawb + (size_t)down[2 * i + 1] * 32);
#pragma unroll
        for (int c = 0; c < 16; ++c) {
            unsigned int p0 = (c < 8 ? r0[2 * c] : r1[2 * c - 16]);
            unsigned int p1 = (c < 8 ? r0[2 * c + 1] : r1[2 * c - 15]);
            float a0 = 0.5f * ((float)__builtin_bit_cast(__bf16, (unsigned short)(p0 & 0xffffu)) +
                               (float)__builtin_bit_cast(__bf16, (unsigned short)(p0 >> 16)));
            float a1 = 0.5f * ((float)__builtin_bit_cast(__bf16, (unsigned short)(p1 & 0xffffu)) +
                               (float)__builtin_bit_cast(__bf16, (unsigned short)(p1 >> 16)));
            unsigned short lo = __builtin_bit_cast(unsigned short, (__bf16)a0);
            unsigned short hi = __builtin_bit_cast(unsigned short, (__bf16)a1);
            w[c] = (unsigned int)lo | ((unsigned int)hi << 16);
        }
    }
    const float* x2b = x2 + (size_t)b * OUT_CH * V2;
#pragma unroll
    for (int c = 0; c < 16; ++c) {
        unsigned short lo = __builtin_bit_cast(unsigned short, (__bf16)x2b[(size_t)(2 * c) * V2 + v]);
        unsigned short hi = __builtin_bit_cast(unsigned short, (__bf16)x2b[(size_t)(2 * c + 1) * V2 + v]);
        w[16 + c] = (unsigned int)lo | ((unsigned int)hi << 16);
    }
    u32x4* yq = (u32x4*)yr;
#pragma unroll
    for (int i = 0; i < 8; ++i) {
        u32x4 q = {w[4 * i], w[4 * i + 1], w[4 * i + 2], w[4 * i + 3]};
        yq[i] = q;
    }
}

// ---------------- conv1: z1 = gather7(y) @ W1^T + b1 ; accumulate BN stats ----------------
__global__ void conv1_gemm(const __bf16* __restrict__ y, const __bf16* __restrict__ SW1,
                           const float* __restrict__ b1, const int* __restrict__ neigh,
                           float* __restrict__ z1, float* __restrict__ sum1,
                           float* __restrict__ sq1) {
    __shared__ float s_sum[32], s_sq[32];
    __shared__ __align__(32) __bf16 wlds[2 * 14 * 512];   // 28672 B of swizzled W1
    __shared__ float ztile[8][512];
    const int b = blockIdx.y;
    const int lane = threadIdx.x & 31;
    const int wave = threadIdx.x >> 5;
    if (threadIdx.x < 32) { s_sum[threadIdx.x] = 0.f; s_sq[threadIdx.x] = 0.f; }
    if (wave == 0) tdm_stage_to_lds(SW1, wlds, 2u * 14u * 512u * 2u);
    __syncthreads();

    const int tile = blockIdx.x * (blockDim.x >> 5) + wave;
    const int nD = lane & 15, g = lane >> 4;
    float p_sum[2] = {0.f, 0.f}, p_sq[2] = {0.f, 0.f};
    if (tile < MT2) {
        const int vert = tile * 16 + (lane & 15);
        const int vc = vert < V2 ? vert : V2 - 1;
        int nb[7];
#pragma unroll
        for (int j = 0; j < 7; ++j) nb[j] = neigh[7 * (size_t)vc + j];
        const __bf16* yb = y + (size_t)b * V2 * 64;
#pragma unroll
        for (int j = 0; j < 7; ++j) __builtin_prefetch(yb + (size_t)nb[j] * 64, 0, 3);

        v8f acc[2] = {};
#pragma unroll
        for (int kc = 0; kc < 14; ++kc) {
            const __bf16* row = yb + (size_t)nb[kc >> 1] * 64 + ((kc & 1) << 5);
            // lane's 16 A elements live in two contiguous 16B chunks of the row
            union { u32x4 q[2]; v16bf a; } u;
            u.q[0] = *(const u32x4*)(row + (g << 3));
            u.q[1] = *(const u32x4*)(row + 16 + (g << 3));
#pragma unroll
            for (int nt = 0; nt < 2; ++nt) {
                v16bf Bt = *(const v16bf*)(wlds + ((size_t)(nt * 14 + kc) * 32 + lane) * 16);
                acc[nt] = __builtin_amdgcn_wmma_f32_16x16x32_bf16(false, u.a, false, Bt,
                                                                  (short)0, acc[nt], false, false);
            }
        }
        float* slab = ztile[wave];
#pragma unroll
        for (int nt = 0; nt < 2; ++nt) {
            float bias = b1[nt * 16 + nD];
#pragma unroll
            for (int r = 0; r < 8; ++r) {
                int vr = tile * 16 + r + 8 * g;
                float zv = acc[nt][r] + bias;
                slab[(r + 8 * g) * 32 + nt * 16 + nD] = zv;
                float zs = (vr < V2) ? zv : 0.f;
                p_sum[nt] += zs;
                p_sq[nt] += zs * zs;
            }
        }
        // contiguous 2048B tile -> global: async LDS->global, 64B (4 x b128) per lane
        wait_ds_done();
        float* dst = z1 + ((size_t)b * V2PAD + (size_t)tile * 16) * 32;
#pragma unroll
        for (int i = 0; i < 4; ++i)
            async_store_b128(dst + lane * 16 + i * 4, slab + lane * 16 + i * 4);
    }
#pragma unroll
    for (int nt = 0; nt < 2; ++nt) {
        atomicAdd(&s_sum[nt * 16 + nD], p_sum[nt]);
        atomicAdd(&s_sq[nt * 16 + nD], p_sq[nt]);
    }
    __syncthreads();
    if (threadIdx.x < 32) {
        atomicAdd(&sum1[b * 32 + threadIdx.x], s_sum[threadIdx.x]);
        atomicAdd(&sq1[b * 32 + threadIdx.x], s_sq[threadIdx.x]);
    }
}

// ---------------- BN finalize: scale/shift per (b, channel) ----------------
__global__ void finalize_stats(const float* __restrict__ sum, const float* __restrict__ sq,
                               const float* __restrict__ gamma, const float* __restrict__ beta,
                               float* __restrict__ scale, float* __restrict__ shift, float invN) {
    int t = blockIdx.x * blockDim.x + threadIdx.x;
    if (t >= NBATCH * 32) return;
    int c = t & 31;
    float mu = sum[t] * invN;
    float var = sq[t] * invN - mu * mu;
    float inv = rsqrtf(var + EPS);
    float sc = gamma[c] * inv;
    scale[t] = sc;
    shift[t] = beta[c] - mu * sc;
}

// ---------------- conv2: BN1+LReLU on the fly during gather, WMMA, stats2 ----------------
__global__ void conv2_gemm(const float* __restrict__ z1, const __bf16* __restrict__ SW2,
                           const float* __restrict__ b2, const int* __restrict__ neigh,
                           const float* __restrict__ scale1, const float* __restrict__ shift1,
                           float* __restrict__ z2, float* __restrict__ sum2,
                           float* __restrict__ sq2) {
    __shared__ float s_sum[32], s_sq[32];
    __shared__ __align__(32) __bf16 wlds[2 * 7 * 512];    // 14336 B of swizzled W2
    __shared__ float ztile[8][512];
    const int b = blockIdx.y;
    const int lane = threadIdx.x & 31;
    const int wave = threadIdx.x >> 5;
    if (threadIdx.x < 32) { s_sum[threadIdx.x] = 0.f; s_sq[threadIdx.x] = 0.f; }
    if (wave == 0) tdm_stage_to_lds(SW2, wlds, 2u * 7u * 512u * 2u);
    __syncthreads();

    const int tile = blockIdx.x * (blockDim.x >> 5) + wave;
    const int nD = lane & 15, g = lane >> 4;
    float p_sum[2] = {0.f, 0.f}, p_sq[2] = {0.f, 0.f};
    if (tile < MT2) {
        const int vert = tile * 16 + (lane & 15);
        const int vc = vert < V2 ? vert : V2 - 1;
        int nb[7];
#pragma unroll
        for (int j = 0; j < 7; ++j) nb[j] = neigh[7 * (size_t)vc + j];
        const float* zb = z1 + (size_t)b * V2PAD * 32;
#pragma unroll
        for (int j = 0; j < 7; ++j) __builtin_prefetch(zb + (size_t)nb[j] * 32, 0, 3);

        float scl[16], shf[16];
#pragma unroll
        for (int e = 0; e < 16; ++e) {
            int cy = a_kmap(e, g);
            scl[e] = scale1[b * 32 + cy];
            shf[e] = shift1[b * 32 + cy];
        }
        v8f acc[2] = {};
#pragma unroll
        for (int kc = 0; kc < 7; ++kc) {
            const float* row = zb + (size_t)nb[kc] * 32;
            float f[16];  // exactly the lane's 16 A-columns, 4 x b128
            *(u32x4*)&f[0]  = *(const u32x4*)(row + 8 * g);
            *(u32x4*)&f[4]  = *(const u32x4*)(row + 8 * g + 4);
            *(u32x4*)&f[8]  = *(const u32x4*)(row + 16 + 8 * g);
            *(u32x4*)&f[12] = *(const u32x4*)(row + 16 + 8 * g + 4);
            v16bf A;
#pragma unroll
            for (int e = 0; e < 16; ++e) {
                float a = scl[e] * f[e] + shf[e];
                a = a >= 0.f ? a : SLOPE * a;
                A[e] = (__bf16)a;
            }
#pragma unroll
            for (int nt = 0; nt < 2; ++nt) {
                v16bf Bt = *(const v16bf*)(wlds + ((size_t)(nt * 7 + kc) * 32 + lane) * 16);
                acc[nt] = __builtin_amdgcn_wmma_f32_16x16x32_bf16(false, A, false, Bt,
                                                                  (short)0, acc[nt], false, false);
            }
        }
        float* slab = ztile[wave];
#pragma unroll
        for (int nt = 0; nt < 2; ++nt) {
            float bias = b2[nt * 16 + nD];
#pragma unroll
            for (int r = 0; r < 8; ++r) {
                int vr = tile * 16 + r + 8 * g;
                float zv = acc[nt][r] + bias;
                slab[(r + 8 * g) * 32 + nt * 16 + nD] = zv;
                float zs = (vr < V2) ? zv : 0.f;
                p_sum[nt] += zs;
                p_sq[nt] += zs * zs;
            }
        }
        wait_ds_done();
        float* dst = z2 + ((size_t)b * V2PAD + (size_t)tile * 16) * 32;
#pragma unroll
        for (int i = 0; i < 4; ++i)
            async_store_b128(dst + lane * 16 + i * 4, slab + lane * 16 + i * 4);
    }
#pragma unroll
    for (int nt = 0; nt < 2; ++nt) {
        atomicAdd(&s_sum[nt * 16 + nD], p_sum[nt]);
        atomicAdd(&s_sq[nt * 16 + nD], p_sq[nt]);
    }
    __syncthreads();
    if (threadIdx.x < 32) {
        atomicAdd(&sum2[b * 32 + threadIdx.x], s_sum[threadIdx.x]);
        atomicAdd(&sq2[b * 32 + threadIdx.x], s_sq[threadIdx.x]);
    }
}

// ---------------- output: BN2 + LReLU + transpose to (B, 32, V2) ----------------
__global__ void finalize_out(const float* __restrict__ z2, const float* __restrict__ scale2,
                             const float* __restrict__ shift2, float* __restrict__ out) {
    const int b = blockIdx.y;
    const int v = blockIdx.x * blockDim.x + threadIdx.x;
    if (v >= V2) return;
    const float* row = z2 + ((size_t)b * V2PAD + v) * 32;
    float f[32];
    const u32x4* rq = (const u32x4*)row;
#pragma unroll
    for (int i = 0; i < 8; ++i) *(u32x4*)&f[i * 4] = rq[i];
#pragma unroll
    for (int c = 0; c < 32; ++c) {
        float val = scale2[b * 32 + c] * f[c] + shift2[b * 32 + c];
        val = val >= 0.f ? val : SLOPE * val;
        out[((size_t)b * 32 + c) * V2 + v] = val;
    }
}

extern "C" void kernel_launch(void* const* d_in, const int* in_sizes, int n_in,
                              void* d_out, int out_size, void* d_ws, size_t ws_size,
                              hipStream_t stream) {
    (void)in_sizes; (void)n_in; (void)out_size; (void)ws_size;
    const float* x1   = (const float*)d_in[0];
    const float* x2   = (const float*)d_in[1];
    const int*   neigh= (const int*)d_in[2];
    const int*   top  = (const int*)d_in[3];
    const int*   down = (const int*)d_in[4];
    const float* W_up = (const float*)d_in[5];
    const float* b_up = (const float*)d_in[6];
    const float* W1   = (const float*)d_in[7];
    const float* b1   = (const float*)d_in[8];
    const float* g1   = (const float*)d_in[9];
    const float* be1  = (const float*)d_in[10];
    const float* W2   = (const float*)d_in[11];
    const float* b2   = (const float*)d_in[12];
    const float* g2   = (const float*)d_in[13];
    const float* be2  = (const float*)d_in[14];
    float* out = (float*)d_out;

    char* ws = (char*)d_ws;
    size_t off = 0;
    auto take = [&](size_t bytes) {
        char* p = ws + off;
        off = (off + bytes + 255) & ~(size_t)255;
        return p;
    };
    __bf16* SWup = (__bf16*)take((size_t)14 * 2 * 512 * 2);
    __bf16* SW1  = (__bf16*)take((size_t)2 * 14 * 512 * 2);
    __bf16* SW2  = (__bf16*)take((size_t)2 * 7 * 512 * 2);
    float* stats = (float*)take(512 * 4);
    float* sum1 = stats, *sq1 = stats + 128, *sum2 = stats + 256, *sq2 = stats + 384;
    float* scale1 = (float*)take(128 * 4);
    float* shift1 = (float*)take(128 * 4);
    float* scale2 = (float*)take(128 * 4);
    float* shift2 = (float*)take(128 * 4);
    __bf16* rawb = (__bf16*)take((size_t)NBATCH * V1PAD * 224 * 2);
    __bf16* ybuf = (__bf16*)take((size_t)NBATCH * V2 * 64 * 2);
    float* z1 = (float*)take((size_t)NBATCH * V2PAD * 32 * 4);
    float* z2 = (float*)take((size_t)NBATCH * V2PAD * 32 * 4);

    zero_stats<<<2, 256, 0, stream>>>(stats, 512);
    swizzle_weights<<<4, 256, 0, stream>>>(W_up, SWup, 14, 2, 64);
    swizzle_weights<<<4, 256, 0, stream>>>(W1, SW1, 2, 14, 448);
    swizzle_weights<<<2, 256, 0, stream>>>(W2, SW2, 2, 7, 224);

    {
        dim3 grid((MT1 + 7) / 8, NBATCH);
        upconv_gemm<<<grid, 256, 0, stream>>>(x1, SWup, b_up, rawb);
    }
    {
        size_t tot = (size_t)NBATCH * V2;
        build_y<<<(int)((tot + 255) / 256), 256, 0, stream>>>(rawb, x2, top, down, ybuf);
    }
    {
        dim3 grid((MT2 + 7) / 8, NBATCH);
        conv1_gemm<<<grid, 256, 0, stream>>>(ybuf, SW1, b1, neigh, z1, sum1, sq1);
    }
    finalize_stats<<<1, 128, 0, stream>>>(sum1, sq1, g1, be1, scale1, shift1, 1.0f / V2);
    {
        dim3 grid((MT2 + 7) / 8, NBATCH);
        conv2_gemm<<<grid, 256, 0, stream>>>(z1, SW2, b2, neigh, scale1, shift1, z2, sum2, sq2);
    }
    finalize_stats<<<1, 128, 0, stream>>>(sum2, sq2, g2, be2, scale2, shift2, 1.0f / V2);
    {
        dim3 grid((V2 + 255) / 256, NBATCH);
        finalize_out<<<grid, 256, 0, stream>>>(z2, scale2, shift2, out);
    }
}
